// MultiHeadCrossAttention_3264175145098
// MI455X (gfx1250) — compile-verified, matrix operand order checked by reading, not run on hardware
//
#include <hip/hip_runtime.h>
#include <hip/hip_bf16.h>

// ---------------------------------------------------------------------------
// Problem constants (reference: B=4, N=2048, C=256, H=4, d=64)
// ---------------------------------------------------------------------------
#define BB 4
#define NN 2048
#define CC 256
#define HH 4
#define DD 64
#define MROWS (BB * NN)          // 8192 token rows
#define ATT_SCALE 0.125f         // 64^-0.5

typedef __attribute__((ext_vector_type(16))) __bf16 v16bf;
typedef __attribute__((ext_vector_type(8)))  float  v8f;
typedef __attribute__((ext_vector_type(4)))  int    v4i;

// ---------------------------------------------------------------------------
// Async global->LDS copy support (gfx1250 GLOBAL_LOAD_ASYNC_TO_LDS_B128,
// tracked by ASYNCcnt). Guarded so we degrade to plain LDS stores if the
// toolchain lacks the builtins.
// ---------------------------------------------------------------------------
#if defined(__has_builtin)
#if __has_builtin(__builtin_amdgcn_global_load_async_to_lds_b128) && \
    __has_builtin(__builtin_amdgcn_s_wait_asynccnt)
#define HAVE_ASYNC_LDS 1
#endif
#endif
#ifndef HAVE_ASYNC_LDS
#define HAVE_ASYNC_LDS 0
#endif

#if HAVE_ASYNC_LDS
#define WAIT_ASYNC(n) __builtin_amdgcn_s_wait_asynccnt(n)
#else
#define WAIT_ASYNC(n) ((void)0)
#endif

// Copy 16 bytes global -> LDS (async when available).
// Builtin signature (from clang diagnostic): param0 is AS1 (global, printed
// `__device__`) pointer-to-int4; param1 is the LDS (AS3) side.
__device__ __forceinline__ void copy16_g2l(__bf16* lds_dst, const __bf16* gsrc) {
#if HAVE_ASYNC_LDS
  __builtin_amdgcn_global_load_async_to_lds_b128(
      (__attribute__((address_space(1))) v4i*)gsrc,
      (__attribute__((address_space(3))) v4i*)lds_dst,
      /*offset=*/0, /*cpol=*/0);
#else
  *(uint4*)lds_dst = *(const uint4*)gsrc;
#endif
}

// ---------------------------------------------------------------------------
// Helpers
// ---------------------------------------------------------------------------
__device__ __forceinline__ v8f zero8() {
  v8f z;
#pragma unroll
  for (int i = 0; i < 8; ++i) z[i] = 0.0f;
  return z;
}

// D = A*B + C  (bf16 inputs, f32 accumulate) -> v_wmma_f32_16x16x32_bf16
__device__ __forceinline__ v8f wmma_bf16(v16bf a, v16bf b, v8f c) {
  return __builtin_amdgcn_wmma_f32_16x16x32_bf16(
      /*neg_a=*/false, a, /*neg_b=*/false, b,
      /*c_mod=*/(short)0, c, /*reuse_a=*/false, /*reuse_b=*/false);
}

// 16x32 bf16 fragment per ISA 7.12.2 (A layout; also used for B-fragments
// where `row` points at this lane's output-column slice with contiguous K):
//   lanes 0-15  : K = k0+{0..7} (elts 0..7),  k0+{16..23} (elts 8..15)
//   lanes 16-31 : K = k0+{8..15},             k0+{24..31}
// Works on global or LDS pointers (generic addressing).
__device__ __forceinline__ v16bf frag_row_bf16(const __bf16* row, int k0, int lane) {
  const int kk = k0 + ((lane >> 4) << 3);
  v16bf r;
#pragma unroll
  for (int j = 0; j < 8; ++j) {
    r[j]     = row[kk + j];
    r[8 + j] = row[kk + 16 + j];
  }
  return r;
}

// ---------------------------------------------------------------------------
// Kernel 0: weight prep — convert Wq/Wkv/Wproj to TRANSPOSED bf16 so that
// GEMM B-fragments are contiguous bf16 loads. One conversion per element.
// ---------------------------------------------------------------------------
__global__ void __launch_bounds__(256)
k_prep_w(const float* __restrict__ Wq, const float* __restrict__ Wkv,
         const float* __restrict__ Wp,
         __bf16* __restrict__ wqT, __bf16* __restrict__ wkvT,
         __bf16* __restrict__ wpT) {
  const int i = blockIdx.x * 256 + threadIdx.x;          // 0 .. 131071
  if (i < CC * CC) {                                     // 65536
    const int k = i >> 8, n = i & (CC - 1);
    wqT[(size_t)n * CC + k] = (__bf16)Wq[i];
    wpT[(size_t)n * CC + k] = (__bf16)Wp[i];
  }
  if (i < CC * 2 * CC) {                                 // 131072
    const int k = i >> 9, n = i & (2 * CC - 1);
    wkvT[(size_t)n * CC + k] = (__bf16)Wkv[i];
  }
}

// ---------------------------------------------------------------------------
// Kernel 1: q = t2_grad @ Wq -> q32 (8192x256 f32) + qbf (B,H,N,d) bf16
//   One block = one 16-row M tile x 8 N tiles (one per wave).
//   A tile staged once to LDS as bf16; all 8 waves build A-frags from LDS.
// ---------------------------------------------------------------------------
__global__ void __launch_bounds__(256)
k_gemm_q(const float* __restrict__ A, const __bf16* __restrict__ WT,
         float* __restrict__ q32, __bf16* __restrict__ qbf) {
  const int lane = threadIdx.x & 31;
  const int wave = threadIdx.x >> 5;
  const int tng  = blockIdx.x & 1;             // N/128 groups (CC/16/8 == 2)
  const int tm   = blockIdx.x >> 1;
  const int tn   = tng * 8 + wave;
  const int n    = tn * 16 + (lane & 15);

  __shared__ __align__(16) __bf16 ldsA[16][CC];
  for (int i = threadIdx.x; i < 16 * CC; i += 256) {
    const int row = i >> 8, col = i & (CC - 1);
    ldsA[row][col] = (__bf16)A[(size_t)(tm * 16 + row) * CC + col];
  }
  __syncthreads();

  v8f acc = zero8();
#pragma unroll
  for (int k0 = 0; k0 < CC; k0 += 32) {
    v16bf a = frag_row_bf16(&ldsA[lane & 15][0], k0, lane);
    v16bf b = frag_row_bf16(WT + (size_t)n * CC, k0, lane);
    acc = wmma_bf16(a, b, acc);
  }
#pragma unroll
  for (int r = 0; r < 8; ++r) {
    const int mm = tm * 16 + r + ((lane >> 4) << 3);
    const float v = acc[r];
    q32[(size_t)mm * CC + n] = v;
    const int b_ = mm >> 11, tok = mm & (NN - 1);
    const int h = n >> 6, e = n & 63;
    qbf[(((size_t)(b_ * HH + h)) * NN + tok) * DD + e] = (__bf16)v;
  }
}

// ---------------------------------------------------------------------------
// Kernel 2: kv = t1 @ Wkv -> kbf (B,H,N,d) bf16, vtbf (B,H,d,N) bf16 (V^T)
// ---------------------------------------------------------------------------
__global__ void __launch_bounds__(256)
k_gemm_kv(const float* __restrict__ A, const __bf16* __restrict__ WT,
          __bf16* __restrict__ kbf, __bf16* __restrict__ vtbf) {
  const int lane = threadIdx.x & 31;
  const int wave = threadIdx.x >> 5;
  const int tng  = blockIdx.x & 3;             // 512/16/8 == 4 N groups
  const int tm   = blockIdx.x >> 2;
  const int tn   = tng * 8 + wave;
  const int n    = tn * 16 + (lane & 15);

  __shared__ __align__(16) __bf16 ldsA[16][CC];
  for (int i = threadIdx.x; i < 16 * CC; i += 256) {
    const int row = i >> 8, col = i & (CC - 1);
    ldsA[row][col] = (__bf16)A[(size_t)(tm * 16 + row) * CC + col];
  }
  __syncthreads();

  v8f acc = zero8();
#pragma unroll
  for (int k0 = 0; k0 < CC; k0 += 32) {
    v16bf a = frag_row_bf16(&ldsA[lane & 15][0], k0, lane);
    v16bf b = frag_row_bf16(WT + (size_t)n * CC, k0, lane);
    acc = wmma_bf16(a, b, acc);
  }
#pragma unroll
  for (int r = 0; r < 8; ++r) {
    const int mm = tm * 16 + r + ((lane >> 4) << 3);
    const int b_ = mm >> 11, tok = mm & (NN - 1);
    const __bf16 bv = (__bf16)acc[r];
    if (n < CC) {                               // K half
      const int h = n >> 6, e = n & 63;
      kbf[(((size_t)(b_ * HH + h)) * NN + tok) * DD + e] = bv;
    } else {                                    // V half -> store transposed
      const int c2 = n - CC;
      const int h = c2 >> 6, e = c2 & 63;
      vtbf[(((size_t)(b_ * HH + h)) * DD + e) * NN + tok] = bv;
    }
  }
}

// ---------------------------------------------------------------------------
// Kernel 3: flash attention per (b,h): xo = softmax(Q K^T * s) V  (B,H,N,d) f32
//   8 waves/block, each wave owns 16 query rows; 32 keys per step.
//   K/V tiles are shared by all waves -> double-buffered async copy into LDS
//   (GLOBAL_LOAD_ASYNC_TO_LDS_B128 + s_wait_asynccnt), fragments via ds_load.
// ---------------------------------------------------------------------------
__global__ void __launch_bounds__(256)
k_attn(const __bf16* __restrict__ qbf, const __bf16* __restrict__ kbf,
       const __bf16* __restrict__ vtbf, float* __restrict__ xo) {
  const int tid  = threadIdx.x;
  const int lane = tid & 31;
  const int wave = tid >> 5;
  const int bh   = blockIdx.x >> 4;            // 16 query blocks per head
  const int qblk = blockIdx.x & 15;
  const int m0   = qblk * 128 + wave * 16;

  const __bf16* Q  = qbf  + (size_t)bh * NN * DD;
  const __bf16* Kp = kbf  + (size_t)bh * NN * DD;
  const __bf16* Vt = vtbf + (size_t)bh * DD * NN;

  __shared__ __align__(16) __bf16 ldsK[2][32][DD];   // 2 x 4KB key tiles
  __shared__ __align__(16) __bf16 ldsV[2][DD][32];   // 2 x 4KB V^T tiles
  __shared__ __align__(16) __bf16 ldsP[8][16][32];   // per-wave P transpose

  // per-thread 16B copy chunks: one for K tile, one for V tile
  const int vrow = tid >> 2;                   // 0..63 (d row)
  const int vseg = (tid & 3) * 8;              // 0,8,16,24 (key seg)

  // stage key-block jb into buffer `buf` (2 async ops per thread)
  auto stage = [&](int buf, int jb) {
    copy16_g2l(&ldsK[buf][0][0] + tid * 8, Kp + (size_t)jb * DD + tid * 8);
    copy16_g2l(&ldsV[buf][vrow][vseg], Vt + (size_t)vrow * NN + jb + vseg);
  };

  const int mrow = m0 + (lane & 15);
  const v16bf aq0 = frag_row_bf16(Q + (size_t)mrow * DD, 0, lane);
  const v16bf aq1 = frag_row_bf16(Q + (size_t)mrow * DD, 32, lane);

  v8f o0 = zero8(), o1 = zero8(), o2 = zero8(), o3 = zero8();
  float mx[8], ls[8];
#pragma unroll
  for (int r = 0; r < 8; ++r) { mx[r] = -3.0e38f; ls[r] = 0.0f; }

  stage(0, 0);
  int cur = 0;

  for (int jb = 0; jb < NN; jb += 32) {
    // prefetch next tile into other buffer, then wait for current tile
    // (async ops complete in order: cnt<=2 means current tile is resident)
    if (jb + 32 < NN) {
      stage(cur ^ 1, jb + 32);
      WAIT_ASYNC(2);
    } else {
      WAIT_ASYNC(0);
    }
    __syncthreads();

    // S = Q K^T over d=64 (two k-steps per 16-key tile)
    const __bf16* kl = &ldsK[cur][lane & 15][0];
    const __bf16* kh = &ldsK[cur][16 + (lane & 15)][0];
    v16bf bk0a = frag_row_bf16(kl, 0, lane);
    v16bf bk0b = frag_row_bf16(kl, 32, lane);
    v16bf bk1a = frag_row_bf16(kh, 0, lane);
    v16bf bk1b = frag_row_bf16(kh, 32, lane);

    v8f s0 = zero8();
    s0 = wmma_bf16(aq0, bk0a, s0);
    s0 = wmma_bf16(aq1, bk0b, s0);
    v8f s1 = zero8();
    s1 = wmma_bf16(aq0, bk1a, s1);
    s1 = wmma_bf16(aq1, bk1b, s1);

    // online softmax per row (rows r+8*(lane>>4); 16-lane halves hold columns)
#pragma unroll
    for (int r = 0; r < 8; ++r) {
      const float a = s0[r] * ATT_SCALE;
      const float b = s1[r] * ATT_SCALE;
      float mloc = fmaxf(a, b);
#pragma unroll
      for (int off = 1; off < 16; off <<= 1)
        mloc = fmaxf(mloc, __shfl_xor(mloc, off));
      const float mnew  = fmaxf(mx[r], mloc);
      const float alpha = __expf(mx[r] - mnew);
      const float p0 = __expf(a - mnew);
      const float p1 = __expf(b - mnew);
      float psum = p0 + p1;
#pragma unroll
      for (int off = 1; off < 16; off <<= 1)
        psum += __shfl_xor(psum, off);
      ls[r] = ls[r] * alpha + psum;
      mx[r] = mnew;
      o0[r] *= alpha; o1[r] *= alpha; o2[r] *= alpha; o3[r] *= alpha;
      const int prow = r + ((lane >> 4) << 3);
      ldsP[wave][prow][lane & 15]        = (__bf16)p0;
      ldsP[wave][prow][16 + (lane & 15)] = (__bf16)p1;
    }

    // P (16x32) as A fragment from per-wave LDS region (same-wave DS is
    // in-order on CDNA5, no barrier needed)
    const v16bf ap = frag_row_bf16(&ldsP[wave][lane & 15][0], 0, lane);

    // O += P @ V : B[k][n] = V[key jb+k][dcol] from LDS V^T tile (contiguous)
    {
      v16bf bv0 = frag_row_bf16(&ldsV[cur][0  + (lane & 15)][0], 0, lane);
      v16bf bv1 = frag_row_bf16(&ldsV[cur][16 + (lane & 15)][0], 0, lane);
      v16bf bv2 = frag_row_bf16(&ldsV[cur][32 + (lane & 15)][0], 0, lane);
      v16bf bv3 = frag_row_bf16(&ldsV[cur][48 + (lane & 15)][0], 0, lane);
      o0 = wmma_bf16(ap, bv0, o0);
      o1 = wmma_bf16(ap, bv1, o1);
      o2 = wmma_bf16(ap, bv2, o2);
      o3 = wmma_bf16(ap, bv3, o3);
    }

    __syncthreads();   // all waves done with `cur` before it is re-staged
    cur ^= 1;
  }

  // finalize: divide by row sums, store xo (B,H,N,d) f32
#pragma unroll
  for (int r = 0; r < 8; ++r) {
    const float inv = 1.0f / ls[r];
    const int m = m0 + r + ((lane >> 4) << 3);
    float* dst = xo + ((size_t)bh * NN + m) * DD;
    dst[0  + (lane & 15)] = o0[r] * inv;
    dst[16 + (lane & 15)] = o1[r] * inv;
    dst[32 + (lane & 15)] = o2[r] * inv;
    dst[48 + (lane & 15)] = o3[r] * inv;
  }
}

// ---------------------------------------------------------------------------
// Kernel 4: lam energy: energy[b][e][f] = sum_m xo(b,m,e)*xo(b,m,f), m over H*N
// ---------------------------------------------------------------------------
__global__ void __launch_bounds__(256)
k_energy(const float* __restrict__ xo, float* __restrict__ energy) {
  const int b = blockIdx.x;
  const float* X = xo + (size_t)b * HH * NN * DD;    // (H*N) rows x 64 ch
  __shared__ __align__(16) float tile[128][DD];
  const int t  = threadIdx.x;
  const int e  = t >> 2;
  const int f0 = (t & 3) << 4;
  float acc[16] = {0.0f};

  for (int mc = 0; mc < HH * NN; mc += 128) {
    const int row  = mc + (t >> 1);
    const int half = (t & 1) * 32;
    const float4* src = (const float4*)(X + (size_t)row * DD + half);
    float4* dst = (float4*)(&tile[t >> 1][half]);
#pragma unroll
    for (int j = 0; j < 8; ++j) dst[j] = src[j];
    __syncthreads();
    for (int ml = 0; ml < 128; ++ml) {
      const float xe = tile[ml][e];
#pragma unroll
      for (int i = 0; i < 16; ++i) acc[i] += xe * tile[ml][f0 + i];
    }
    __syncthreads();
  }
#pragma unroll
  for (int i = 0; i < 16; ++i)
    energy[((size_t)b * DD + e) * DD + f0 + i] = acc[i];
}

// ---------------------------------------------------------------------------
// Kernel 5: row softmax of 64x64 energy per batch
// ---------------------------------------------------------------------------
__global__ void k_softmax64(const float* __restrict__ energy,
                            float* __restrict__ attnE) {
  const int b = blockIdx.x;
  const int e = threadIdx.x;                   // 64 threads
  const float* row = energy + ((size_t)b * DD + e) * DD;
  float mx = -3.0e38f;
  for (int f = 0; f < DD; ++f) mx = fmaxf(mx, row[f]);
  float s = 0.0f;
  for (int f = 0; f < DD; ++f) s += __expf(row[f] - mx);
  const float inv = 1.0f / s;
  float* out = attnE + ((size_t)b * DD + e) * DD;
  for (int f = 0; f < DD; ++f) out[f] = __expf(row[f] - mx) * inv;
}

// ---------------------------------------------------------------------------
// Kernel 6: combine: comb = bf16( gamma * (attnE @ xo) + xo + q + t2_grad )
//   one block per token row (b,n); thread = output channel c = h*64+e
// ---------------------------------------------------------------------------
__global__ void __launch_bounds__(256)
k_combine(const float* __restrict__ xo, const float* __restrict__ attnE,
          const float* __restrict__ q32, const float* __restrict__ t2,
          const float* __restrict__ gamma, __bf16* __restrict__ comb) {
  const int rowid = blockIdx.x;                // b*N + n
  const int b = rowid >> 11, n = rowid & (NN - 1);
  const int c = threadIdx.x;
  const int h = c >> 6, e = c & 63;

  __shared__ float sE[DD * DD];
  __shared__ float sx[CC];
  sx[c] = xo[(((size_t)(b * HH + h)) * NN + n) * DD + e];
  const float* Eb = attnE + (size_t)b * DD * DD;
  for (int i = c; i < DD * DD; i += 256) sE[i] = Eb[i];
  __syncthreads();

  const float g = gamma[0];
  float dot = 0.0f;
#pragma unroll 8
  for (int f = 0; f < DD; ++f) dot += sE[e * DD + f] * sx[(h << 6) + f];

  const float val = g * dot + sx[c]
                  + q32[(size_t)rowid * CC + c]
                  + t2[(size_t)rowid * CC + c];
  comb[(size_t)rowid * CC + c] = (__bf16)val;
}

// ---------------------------------------------------------------------------
// Kernel 7: out = comb @ Wproj + bproj  (bf16 A and B^T, f32 out)
// ---------------------------------------------------------------------------
__global__ void __launch_bounds__(256)
k_gemm_proj(const __bf16* __restrict__ A, const __bf16* __restrict__ WT,
            const float* __restrict__ bias, float* __restrict__ out) {
  const int lane = threadIdx.x & 31;
  const int wave = threadIdx.x >> 5;
  const int tile = blockIdx.x * 8 + wave;
  const int tm = tile >> 4;
  const int tn = tile & 15;
  const int m = tm * 16 + (lane & 15);
  const int n = tn * 16 + (lane & 15);

  v8f acc = zero8();
#pragma unroll
  for (int k0 = 0; k0 < CC; k0 += 32) {
    v16bf a = frag_row_bf16(A + (size_t)m * CC, k0, lane);
    v16bf b = frag_row_bf16(WT + (size_t)n * CC, k0, lane);
    acc = wmma_bf16(a, b, acc);
  }
  const float bn = bias[n];
#pragma unroll
  for (int r = 0; r < 8; ++r) {
    const int mm = tm * 16 + r + ((lane >> 4) << 3);
    out[(size_t)mm * CC + n] = acc[r] + bn;
  }
}

// ---------------------------------------------------------------------------
// Launch
// ---------------------------------------------------------------------------
static inline unsigned char* align_up(unsigned char* p, size_t a) {
  return (unsigned char*)(((uintptr_t)p + (a - 1)) & ~(uintptr_t)(a - 1));
}

extern "C" void kernel_launch(void* const* d_in, const int* in_sizes, int n_in,
                              void* d_out, int out_size, void* d_ws, size_t ws_size,
                              hipStream_t stream) {
  (void)in_sizes; (void)n_in; (void)out_size; (void)ws_size;
  const float* t2    = (const float*)d_in[0];   // (B,N,C)
  const float* t1    = (const float*)d_in[1];   // (B,N,C)
  const float* Wq    = (const float*)d_in[2];   // (C,C)
  const float* Wkv   = (const float*)d_in[3];   // (C,2C)
  const float* Wproj = (const float*)d_in[4];   // (C,C)
  const float* bproj = (const float*)d_in[5];   // (C,)
  const float* gamma = (const float*)d_in[6];   // (1,)
  float* out = (float*)d_out;

  unsigned char* p = (unsigned char*)d_ws;
  float*  q32    = (float*)align_up(p, 256);   p = (unsigned char*)(q32 + (size_t)MROWS * CC);
  __bf16* qbf    = (__bf16*)align_up(p, 256);  p = (unsigned char*)(qbf + (size_t)MROWS * CC);
  __bf16* kbf    = (__bf16*)align_up(p, 256);  p = (unsigned char*)(kbf + (size_t)MROWS * CC);
  __bf16* vtbf   = (__bf16*)align_up(p, 256);  p = (unsigned char*)(vtbf + (size_t)MROWS * CC);
  float*  xo     = (float*)align_up(p, 256);   p = (unsigned char*)(xo + (size_t)BB * HH * NN * DD);
  float*  energy = (float*)align_up(p, 256);   p = (unsigned char*)(energy + (size_t)BB * DD * DD);
  float*  attnE  = (float*)align_up(p, 256);   p = (unsigned char*)(attnE + (size_t)BB * DD * DD);
  __bf16* comb   = (__bf16*)align_up(p, 256);  p = (unsigned char*)(comb + (size_t)MROWS * CC);
  __bf16* wqT    = (__bf16*)align_up(p, 256);  p = (unsigned char*)(wqT + (size_t)CC * CC);
  __bf16* wkvT   = (__bf16*)align_up(p, 256);  p = (unsigned char*)(wkvT + (size_t)CC * 2 * CC);
  __bf16* wpT    = (__bf16*)align_up(p, 256);

  // 0) convert + transpose weights to bf16 once
  k_prep_w<<<(CC * 2 * CC) / 256, 256, 0, stream>>>(Wq, Wkv, Wproj, wqT, wkvT, wpT);
  // 1) q projection (LDS-staged A tile)
  k_gemm_q<<<(MROWS / 16) * 2, 256, 0, stream>>>(t2, wqT, q32, qbf);
  // 2) kv projection (K bf16 row-major, V transposed bf16)
  k_gemm_kv<<<(MROWS / 16) * 4, 256, 0, stream>>>(t1, wkvT, kbf, vtbf);
  // 3) flash attention (async double-buffered K/V tiles in LDS)
  k_attn<<<BB * HH * (NN / 128), 256, 0, stream>>>(qbf, kbf, vtbf, xo);
  // 4) lam energy + softmax
  k_energy<<<BB, 256, 0, stream>>>(xo, energy);
  k_softmax64<<<BB, DD, 0, stream>>>(energy, attnE);
  // 5) combine residuals -> bf16
  k_combine<<<MROWS, CC, 0, stream>>>(xo, attnE, q32, t2, gamma, comb);
  // 6) output projection + bias
  k_gemm_proj<<<(MROWS / 16) * (CC / 16) / 8, 256, 0, stream>>>(comb, wpT, bproj, out);
}
